// DAWN_87677462380612
// MI455X (gfx1250) — matrix operand order repaired and checked by reading, not compiled, not run on hardware
//
#include <hip/hip_runtime.h>
#include <math.h>

typedef __attribute__((ext_vector_type(16))) _Float16 v16h;
typedef __attribute__((ext_vector_type(4)))  _Float16 h4;
typedef __attribute__((ext_vector_type(8)))  float    v8f;

// Model dims (fixed by the reference)
#define BB 2
#define SS 1024
#define DD 512
#define DFF 2048
#define LL 4
#define HH 8
#define NNEUR 512
#define TOPK 16
#define NBASIS 8
#define RR 64
#define VV 32000
#define MTOK (BB*SS)   // 2048 tokens

// ---------------------------------------------------------------------------
// Generic WMMA GEMM: C[M,N] (fp32) = op(A[M,K] @ W) with f16 fragments.
//   TRANSW==0: W is [K,N] row-major.  TRANSW==1: W is [N,K] row-major (A@W^T).
//   bias:      optional per-column bias (N).
//   rowscale:  optional per-row scalar (stride rs_stride) applied to the product.
//   accum:     C += result (else C = result).
// Block: 256 threads = 8 waves (4 M x 2 N), tile 128x64, K-chunk 32.
// Each wave owns a 32x32 subtile: 2 A-frags x 2 B-frags -> 4 WMMAs/chunk,
// software-pipelined 128-bit staging with L2 prefetch.
// All dims are multiples of the tile in this model (M=2048; N,K in
// {64,512,2048,32000} are multiples of 64/32).
// ---------------------------------------------------------------------------
template <int TRANSW>
__global__ __launch_bounds__(256)
void gemm_wmma_f16(const float* __restrict__ A, const float* __restrict__ W,
                   const float* __restrict__ bias, const float* __restrict__ rowscale,
                   int rs_stride, float* __restrict__ C,
                   int M, int N, int Kd, int accum)
{
    __shared__ __align__(16) _Float16 As[128][40];  // 128 x 32 (+pad, 80B rows)
    __shared__ __align__(16) _Float16 Bs[32][72];   //  32 x 64 (+pad, 144B rows)

    const int bn = blockIdx.x * 64;
    const int bm = blockIdx.y * 128;
    const int tid  = threadIdx.x;
    const int wave = tid >> 5;
    const int lane = tid & 31;
    const int Moff = (wave >> 1) * 32;   // 4 wave-rows of 32
    const int Noff = (wave & 1) * 32;    // 2 wave-cols of 32

    // Staging coordinates: 4x float4 per thread for A, 2x float4 for W.
    const int ar0 = tid >> 3;            // 0..31 (A rows ar0 + {0,32,64,96})
    const int ac4 = (tid & 7) * 4;       // 0..28
    const int br0 = tid >> 4;            // 0..15 (B rows br0, br0+16)
    const int bc4 = (tid & 15) * 4;      // 0..60
    const int tbc = tid & 63;            // TRANSW: column
    const int tbr = (tid >> 6) * 8;      // TRANSW: k-run of 8

    float4 ra0, ra1, ra2, ra3, rb0, rb1;

    auto loadT = [&](int k0) {
        ra0 = *(const float4*)&A[(size_t)(bm + ar0)      * Kd + k0 + ac4];
        ra1 = *(const float4*)&A[(size_t)(bm + ar0 + 32) * Kd + k0 + ac4];
        ra2 = *(const float4*)&A[(size_t)(bm + ar0 + 64) * Kd + k0 + ac4];
        ra3 = *(const float4*)&A[(size_t)(bm + ar0 + 96) * Kd + k0 + ac4];
        if (TRANSW) {
            rb0 = *(const float4*)&W[(size_t)(bn + tbc) * Kd + k0 + tbr];
            rb1 = *(const float4*)&W[(size_t)(bn + tbc) * Kd + k0 + tbr + 4];
        } else {
            rb0 = *(const float4*)&W[(size_t)(k0 + br0)      * N + bn + bc4];
            rb1 = *(const float4*)&W[(size_t)(k0 + br0 + 16) * N + bn + bc4];
        }
    };
    auto cvt4 = [](float4 f) {
        h4 h; h[0] = (_Float16)f.x; h[1] = (_Float16)f.y;
              h[2] = (_Float16)f.z; h[3] = (_Float16)f.w; return h;
    };
    auto storeT = [&]() {
        *(h4*)&As[ar0][ac4]      = cvt4(ra0);
        *(h4*)&As[ar0 + 32][ac4] = cvt4(ra1);
        *(h4*)&As[ar0 + 64][ac4] = cvt4(ra2);
        *(h4*)&As[ar0 + 96][ac4] = cvt4(ra3);
        if (TRANSW) {
            h4 h0 = cvt4(rb0), h1 = cvt4(rb1);
            Bs[tbr + 0][tbc] = h0[0]; Bs[tbr + 1][tbc] = h0[1];
            Bs[tbr + 2][tbc] = h0[2]; Bs[tbr + 3][tbc] = h0[3];
            Bs[tbr + 4][tbc] = h1[0]; Bs[tbr + 5][tbc] = h1[1];
            Bs[tbr + 6][tbc] = h1[2]; Bs[tbr + 7][tbc] = h1[3];
        } else {
            *(h4*)&Bs[br0][bc4]      = cvt4(rb0);
            *(h4*)&Bs[br0 + 16][bc4] = cvt4(rb1);
        }
    };

    v8f c00 = {}, c01 = {}, c10 = {}, c11 = {};
    const int m0  = Moff + (lane & 15);
    const int kb8 = (lane >> 4) * 8;     // A-frag K offset for upper half-wave

    loadT(0);
    for (int k0 = 0; k0 < Kd; k0 += 32) {
        storeT();
        __syncthreads();
        if (k0 + 32 < Kd) loadT(k0 + 32);          // overlap with compute
        if (k0 + 64 < Kd) {                        // L2 prefetch two chunks out
            __builtin_prefetch(&A[(size_t)(bm + ar0) * Kd + k0 + 64 + ac4], 0, 1);
            if (TRANSW)
                __builtin_prefetch(&W[(size_t)(bn + tbc) * Kd + k0 + 64 + tbr], 0, 1);
            else
                __builtin_prefetch(&W[(size_t)(k0 + 64 + br0) * N + bn + bc4], 0, 1);
        }

        // A fragments: lane = M row; VGPR halves follow ISA 16-bit A 16x32 layout
        v16h a0, a1;
        #pragma unroll
        for (int i = 0; i < 8; ++i) {
            int kk = ((i < 4) ? (2 * i) : (16 + 2 * (i - 4))) + kb8;
            a0[2 * i]     = As[m0][kk];
            a0[2 * i + 1] = As[m0][kk + 1];
            a1[2 * i]     = As[m0 + 16][kk];
            a1[2 * i + 1] = As[m0 + 16][kk + 1];
        }
        // B fragments: lane = N column; elements = K in same order as A
        v16h b0, b1;
        #pragma unroll
        for (int i = 0; i < 8; ++i) {
            int kk = ((i < 4) ? (2 * i) : (16 + 2 * (i - 4))) + kb8;
            b0[2 * i]     = Bs[kk][Noff + (lane & 15)];
            b0[2 * i + 1] = Bs[kk + 1][Noff + (lane & 15)];
            b1[2 * i]     = Bs[kk][Noff + 16 + (lane & 15)];
            b1[2 * i + 1] = Bs[kk + 1][Noff + 16 + (lane & 15)];
        }
        c00 = __builtin_amdgcn_wmma_f32_16x16x32_f16(false, a0, false, b0, (short)0, c00, false, false);
        c01 = __builtin_amdgcn_wmma_f32_16x16x32_f16(false, a0, false, b1, (short)0, c01, false, false);
        c10 = __builtin_amdgcn_wmma_f32_16x16x32_f16(false, a1, false, b0, (short)0, c10, false, false);
        c11 = __builtin_amdgcn_wmma_f32_16x16x32_f16(false, a1, false, b1, (short)0, c11, false, false);
        __syncthreads();
    }

    // Epilogue: C-frag layout -> lane = N col, VGPR r = M row (+8 for lanes>=16)
    const int hi   = lane >> 4;
    const int col0 = bn + Noff + (lane & 15);
    #pragma unroll
    for (int r = 0; r < 8; ++r) {
        int row0 = bm + Moff + r + 8 * hi;
        int row1 = row0 + 16;
        float v00 = c00[r], v01 = c01[r], v10 = c10[r], v11 = c11[r];
        if (bias) {
            float bz0 = bias[col0], bz1 = bias[col0 + 16];
            v00 += bz0; v01 += bz1; v10 += bz0; v11 += bz1;
        }
        if (rowscale) {
            float s0 = rowscale[(size_t)row0 * rs_stride];
            float s1 = rowscale[(size_t)row1 * rs_stride];
            v00 *= s0; v01 *= s0; v10 *= s1; v11 *= s1;
        }
        size_t o0 = (size_t)row0 * N + col0;
        size_t o1 = (size_t)row1 * N + col0;
        if (accum) { C[o0] += v00; C[o0 + 16] += v01; C[o1] += v10; C[o1 + 16] += v11; }
        else       { C[o0]  = v00; C[o0 + 16]  = v01; C[o1]  = v10; C[o1 + 16]  = v11; }
    }
}

// ---------------------------------------------------------------------------
// Flash-style causal attention with WMMA. One wave per (b, h, 16-query tile).
// q/k/v layout: [B*S, D] with head hd occupying columns hd*64 .. hd*64+63.
// ---------------------------------------------------------------------------
__global__ __launch_bounds__(32)
void attn_wmma(const float* __restrict__ q, const float* __restrict__ k,
               const float* __restrict__ v, float* __restrict__ ctx)
{
    const int qt = blockIdx.x;          // query tile (S/16)
    const int hd = blockIdx.y;          // head
    const int bb = blockIdx.z;          // batch
    const int lane = threadIdx.x;
    const int q0 = qt * 16;

    const float* qb_ = q + (size_t)bb * SS * DD + hd * 64;
    const float* kb_ = k + (size_t)bb * SS * DD + hd * 64;
    const float* vb_ = v + (size_t)bb * SS * DD + hd * 64;

    const int m   = lane & 15;
    const int hi  = lane >> 4;
    const int kb8 = hi * 8;
    const int c   = lane & 15;

    // Q A-fragments for dh=64 split into two K=32 chunks; fold 1/sqrt(64)
    v16h aq0, aq1;
    #pragma unroll
    for (int i = 0; i < 8; ++i) {
        int kk = ((i < 4) ? (2 * i) : (16 + 2 * (i - 4))) + kb8;
        const float* qr = qb_ + (size_t)(q0 + m) * DD;
        aq0[2 * i]     = (_Float16)(0.125f * qr[kk]);
        aq0[2 * i + 1] = (_Float16)(0.125f * qr[kk + 1]);
        aq1[2 * i]     = (_Float16)(0.125f * qr[32 + kk]);
        aq1[2 * i + 1] = (_Float16)(0.125f * qr[32 + kk + 1]);
    }

    v8f o0 = {}, o1 = {}, o2 = {}, o3 = {};
    float mrow[8], lrow[8];
    #pragma unroll
    for (int r = 0; r < 8; ++r) { mrow[r] = -1e30f; lrow[r] = 0.f; }

    __shared__ __align__(16) _Float16 Ps[16][36];

    for (int kc = 0; kc <= q0; kc += 32) {   // causal key chunks of 32
        // K^T B-fragments: lane = key, elements = feature in same order as A
        v16h bk0, bk1, bk0b, bk1b;
        #pragma unroll
        for (int i = 0; i < 8; ++i) {
            int kk = ((i < 4) ? (2 * i) : (16 + 2 * (i - 4))) + kb8;
            const float* kr0 = kb_ + (size_t)(kc + c) * DD;
            const float* kr1 = kb_ + (size_t)(kc + 16 + c) * DD;
            bk0[2 * i]      = (_Float16)kr0[kk];
            bk0[2 * i + 1]  = (_Float16)kr0[kk + 1];
            bk1[2 * i]      = (_Float16)kr0[32 + kk];
            bk1[2 * i + 1]  = (_Float16)kr0[32 + kk + 1];
            bk0b[2 * i]     = (_Float16)kr1[kk];
            bk0b[2 * i + 1] = (_Float16)kr1[kk + 1];
            bk1b[2 * i]     = (_Float16)kr1[32 + kk];
            bk1b[2 * i + 1] = (_Float16)kr1[32 + kk + 1];
        }
        v8f s0 = {}, s1 = {};
        s0 = __builtin_amdgcn_wmma_f32_16x16x32_f16(false, aq0, false, bk0,  (short)0, s0, false, false);
        s0 = __builtin_amdgcn_wmma_f32_16x16x32_f16(false, aq1, false, bk1,  (short)0, s0, false, false);
        s1 = __builtin_amdgcn_wmma_f32_16x16x32_f16(false, aq0, false, bk0b, (short)0, s1, false, false);
        s1 = __builtin_amdgcn_wmma_f32_16x16x32_f16(false, aq1, false, bk1b, (short)0, s1, false, false);

        // Online softmax per row (rows live across 16-lane groups)
        float p0[8], p1[8], rscale[8];
        #pragma unroll
        for (int r = 0; r < 8; ++r) {
            int row = r + 8 * hi, qg = q0 + row;
            float v0 = (kc + c      <= qg) ? s0[r] : -1e30f;
            float v1 = (kc + 16 + c <= qg) ? s1[r] : -1e30f;
            float t = fmaxf(v0, v1);
            #pragma unroll
            for (int off = 1; off < 16; off <<= 1) t = fmaxf(t, __shfl_xor(t, off, 32));
            float mnew = fmaxf(mrow[r], t);
            float sc = __expf(mrow[r] - mnew);
            mrow[r] = mnew;
            v0 = __expf(v0 - mnew);
            v1 = __expf(v1 - mnew);
            float rs = v0 + v1;
            #pragma unroll
            for (int off = 1; off < 16; off <<= 1) rs += __shfl_xor(rs, off, 32);
            lrow[r] = lrow[r] * sc + rs;
            rscale[r] = sc; p0[r] = v0; p1[r] = v1;
        }
        #pragma unroll
        for (int r = 0; r < 8; ++r) {
            o0[r] *= rscale[r]; o1[r] *= rscale[r];
            o2[r] *= rscale[r]; o3[r] *= rscale[r];
        }
        // Transpose P (C-layout, lane=col) to A-layout (lane=row) via LDS
        #pragma unroll
        for (int r = 0; r < 8; ++r) {
            Ps[r + 8 * hi][c]      = (_Float16)p0[r];
            Ps[r + 8 * hi][16 + c] = (_Float16)p1[r];
        }
        __syncthreads();
        v16h pa;
        #pragma unroll
        for (int i = 0; i < 8; ++i) {
            int kk = ((i < 4) ? (2 * i) : (16 + 2 * (i - 4))) + kb8;
            pa[2 * i]     = Ps[m][kk];
            pa[2 * i + 1] = Ps[m][kk + 1];
        }
        // V B-fragments: lane = feature col, elements = key in same order as P; O += P@V
        #pragma unroll
        for (int f = 0; f < 4; ++f) {
            v16h bv;
            const float* vr = vb_ + (size_t)kc * DD + f * 16 + c;
            #pragma unroll
            for (int i = 0; i < 8; ++i) {
                int kk = ((i < 4) ? (2 * i) : (16 + 2 * (i - 4))) + kb8;
                bv[2 * i]     = (_Float16)vr[(size_t)kk * DD];
                bv[2 * i + 1] = (_Float16)vr[(size_t)(kk + 1) * DD];
            }
            if (f == 0) o0 = __builtin_amdgcn_wmma_f32_16x16x32_f16(false, pa, false, bv, (short)0, o0, false, false);
            if (f == 1) o1 = __builtin_amdgcn_wmma_f32_16x16x32_f16(false, pa, false, bv, (short)0, o1, false, false);
            if (f == 2) o2 = __builtin_amdgcn_wmma_f32_16x16x32_f16(false, pa, false, bv, (short)0, o2, false, false);
            if (f == 3) o3 = __builtin_amdgcn_wmma_f32_16x16x32_f16(false, pa, false, bv, (short)0, o3, false, false);
        }
        __syncthreads();
    }

    // Write ctx = O / l
    #pragma unroll
    for (int r = 0; r < 8; ++r) {
        int row = q0 + r + 8 * hi;
        float inv = 1.0f / lrow[r];
        float* cr = ctx + (size_t)(bb * SS + row) * DD + hd * 64;
        cr[0 * 16 + c] = o0[r] * inv;
        cr[1 * 16 + c] = o1[r] * inv;
        cr[2 * 16 + c] = o2[r] * inv;
        cr[3 * 16 + c] = o3[r] * inv;
    }
}

// ---------------------------------------------------------------------------
// Small VALU kernels
// ---------------------------------------------------------------------------
__global__ __launch_bounds__(256)
void embed_kernel(const int* __restrict__ ids, const float* __restrict__ te,
                  const float* __restrict__ pe, float* __restrict__ x)
{
    int i = blockIdx.x * 256 + threadIdx.x;           // over MTOK*DD
    int tok = i / DD, col = i % DD;
    int s = tok % SS;
    x[i] = te[(size_t)ids[tok] * DD + col] + pe[(size_t)s * DD + col];
}

__global__ __launch_bounds__(256)
void ln_kernel(const float* __restrict__ x, const float* __restrict__ sc,
               const float* __restrict__ bi, float* __restrict__ y)
{
    int t = blockIdx.x, tid = threadIdx.x;
    __shared__ float red[256];
    const float* xt = x + (size_t)t * DD;
    float v0 = xt[tid], v1 = xt[tid + 256];
    red[tid] = v0 + v1; __syncthreads();
    for (int s = 128; s; s >>= 1) { if (tid < s) red[tid] += red[tid + s]; __syncthreads(); }
    float mean = red[0] * (1.0f / DD); __syncthreads();
    float d0 = v0 - mean, d1 = v1 - mean;
    red[tid] = d0 * d0 + d1 * d1; __syncthreads();
    for (int s = 128; s; s >>= 1) { if (tid < s) red[tid] += red[tid + s]; __syncthreads(); }
    float rstd = rsqrtf(red[0] * (1.0f / DD) + 1e-5f);
    float* yt = y + (size_t)t * DD;
    yt[tid]       = d0 * rstd * sc[tid]       + bi[tid];
    yt[tid + 256] = d1 * rstd * sc[tid + 256] + bi[tid + 256];
}

__global__ __launch_bounds__(256)
void routing_kernel(float* __restrict__ x, const float* __restrict__ n1,
                    const float* __restrict__ ctx, const float* __restrict__ ts,
                    const float* __restrict__ cs, const float* __restrict__ gw,
                    const float* __restrict__ gb, const float* __restrict__ nemb,
                    const float* __restrict__ nc, float* __restrict__ tc)
{
    int t = blockIdx.x, tid = threadIdx.x;
    __shared__ float red[256];
    __shared__ int   redi[256];
    __shared__ float sc[NNEUR];
    __shared__ int   topi[TOPK];
    __shared__ float topv[TOPK];
    __shared__ float wk[TOPK];

    const float* n1t = n1 + (size_t)t * DD;
    const float* ctt = ctx + (size_t)t * DD;

    // gate logits: concat(n1, ctx) @ gw[2D,2] + gb
    float g0 = 0.f, g1 = 0.f;
    for (int i = tid; i < DD; i += 256) {
        g0 += n1t[i] * gw[i * 2]     + ctt[i] * gw[(DD + i) * 2];
        g1 += n1t[i] * gw[i * 2 + 1] + ctt[i] * gw[(DD + i) * 2 + 1];
    }
    red[tid] = g0; __syncthreads();
    for (int s = 128; s; s >>= 1) { if (tid < s) red[tid] += red[tid + s]; __syncthreads(); }
    g0 = red[0] + gb[0]; __syncthreads();
    red[tid] = g1; __syncthreads();
    for (int s = 128; s; s >>= 1) { if (tid < s) red[tid] += red[tid + s]; __syncthreads(); }
    g1 = red[0] + gb[1]; __syncthreads();
    float mg = fmaxf(g0, g1);
    float e0 = __expf(g0 - mg), e1 = __expf(g1 - mg);
    float ga = e0 / (e0 + e1), gcx = e1 / (e0 + e1);

    for (int j = tid; j < NNEUR; j += 256)
        sc[j] = ga * ts[(size_t)t * NNEUR + j] + gcx * cs[(size_t)t * NNEUR + j];
    __syncthreads();

    // top-16 by iterative argmax
    for (int kk = 0; kk < TOPK; ++kk) {
        float bv = -1e30f; int bi2 = 0;
        for (int j = tid; j < NNEUR; j += 256)
            if (sc[j] > bv) { bv = sc[j]; bi2 = j; }
        red[tid] = bv; redi[tid] = bi2; __syncthreads();
        for (int s = 128; s; s >>= 1) {
            if (tid < s && red[tid + s] > red[tid]) { red[tid] = red[tid + s]; redi[tid] = redi[tid + s]; }
            __syncthreads();
        }
        if (tid == 0) { topi[kk] = redi[0]; topv[kk] = red[0]; sc[redi[0]] = -1e30f; }
        __syncthreads();
    }
    if (tid == 0) {
        float mx = topv[0], ssum = 0.f;
        for (int kk = 0; kk < TOPK; ++kk) { wk[kk] = __expf(topv[kk] - mx); ssum += wk[kk]; }
        float inv = 1.0f / ssum;
        for (int kk = 0; kk < TOPK; ++kk) wk[kk] *= inv;
    }
    __syncthreads();

    // x += sum_k w_k * nemb[topk]
    for (int col = tid; col < DD; col += 256) {
        float acc = 0.f;
        #pragma unroll
        for (int kk = 0; kk < TOPK; ++kk) acc += wk[kk] * nemb[(size_t)topi[kk] * DD + col];
        x[(size_t)t * DD + col] += acc;
    }
    // tc = sum_k w_k * nc[topk]
    if (tid < NBASIS) {
        float acc = 0.f;
        #pragma unroll
        for (int kk = 0; kk < TOPK; ++kk) acc += wk[kk] * nc[(size_t)topi[kk] * NBASIS + tid];
        tc[(size_t)t * NBASIS + tid] = acc;
    }
}

__global__ __launch_bounds__(256)
void gelu_kernel(float* __restrict__ h)
{
    int i = blockIdx.x * 256 + threadIdx.x;
    float v = h[i];
    h[i] = 0.5f * v * (1.0f + erff(v * 0.70710678118654752f));
}

// ---------------------------------------------------------------------------
// Host-side orchestration (graph-capture safe: only kernel launches on stream)
// Workspace: ~55.1 MB fp32.
// ---------------------------------------------------------------------------
extern "C" void kernel_launch(void* const* d_in, const int* in_sizes, int n_in,
                              void* d_out, int out_size, void* d_ws, size_t ws_size,
                              hipStream_t stream) {
    const int*   ids  = (const int*)  d_in[0];
    const float* te   = (const float*)d_in[1];
    const float* pe   = (const float*)d_in[2];
    const float* qw   = (const float*)d_in[3];
    const float* qb   = (const float*)d_in[4];
    const float* kw   = (const float*)d_in[5];
    const float* kbi  = (const float*)d_in[6];
    const float* vw   = (const float*)d_in[7];
    const float* vbi  = (const float*)d_in[8];
    const float* nemb = (const float*)d_in[9];
    const float* gw   = (const float*)d_in[10];
    const float* gb   = (const float*)d_in[11];
    const float* n1s  = (const float*)d_in[12];
    const float* n1b  = (const float*)d_in[13];
    const float* n2s  = (const float*)d_in[14];
    const float* n2b  = (const float*)d_in[15];
    const float* bA   = (const float*)d_in[16];
    const float* bBm  = (const float*)d_in[17];
    const float* nc   = (const float*)d_in[18];
    const float* dww  = (const float*)d_in[19];
    const float* dwb  = (const float*)d_in[20];
    const float* nfs  = (const float*)d_in[21];
    const float* nfb  = (const float*)d_in[22];
    float* out = (float*)d_out;

    float* p = (float*)d_ws;
    float* x   = p; p += (size_t)MTOK * DD;
    float* n1  = p; p += (size_t)MTOK * DD;
    float* n2  = p; p += (size_t)MTOK * DD;
    float* qv  = p; p += (size_t)MTOK * DD;
    float* kv  = p; p += (size_t)MTOK * DD;
    float* vv  = p; p += (size_t)MTOK * DD;
    float* ctx = p; p += (size_t)MTOK * DD;
    float* ts  = p; p += (size_t)MTOK * NNEUR;
    float* cs  = p; p += (size_t)MTOK * NNEUR;
    float* hb  = p; p += (size_t)MTOK * RR;
    float* hf  = p; p += (size_t)MTOK * DFF;
    float* tc  = p; p += (size_t)MTOK * NBASIS;

    embed_kernel<<<(MTOK * DD) / 256, 256, 0, stream>>>(ids, te, pe, x);

    for (int l = 0; l < LL; ++l) {
        const float* nemb_l = nemb + (size_t)l * NNEUR * DD;

        ln_kernel<<<MTOK, 256, 0, stream>>>(x, n1s + l * DD, n1b + l * DD, n1);

        dim3 gDD(DD / 64, MTOK / 128);
        gemm_wmma_f16<0><<<gDD, 256, 0, stream>>>(n1, qw + (size_t)l * DD * DD, qb + l * DD,
                                                  nullptr, 0, qv, MTOK, DD, DD, 0);
        gemm_wmma_f16<0><<<gDD, 256, 0, stream>>>(n1, kw + (size_t)l * DD * DD, kbi + l * DD,
                                                  nullptr, 0, kv, MTOK, DD, DD, 0);
        gemm_wmma_f16<0><<<gDD, 256, 0, stream>>>(n1, vw + (size_t)l * DD * DD, vbi + l * DD,
                                                  nullptr, 0, vv, MTOK, DD, DD, 0);

        attn_wmma<<<dim3(SS / 16, HH, BB), 32, 0, stream>>>(qv, kv, vv, ctx);

        dim3 gNN(NNEUR / 64, MTOK / 128);
        gemm_wmma_f16<1><<<gNN, 256, 0, stream>>>(n1,  nemb_l, nullptr, nullptr, 0, ts,
                                                  MTOK, NNEUR, DD, 0);
        gemm_wmma_f16<1><<<gNN, 256, 0, stream>>>(ctx, nemb_l, nullptr, nullptr, 0, cs,
                                                  MTOK, NNEUR, DD, 0);

        routing_kernel<<<MTOK, 256, 0, stream>>>(x, n1, ctx, ts, cs,
                                                 gw + (size_t)l * 2 * DD * 2, gb + l * 2,
                                                 nemb_l, nc + (size_t)l * NNEUR * NBASIS, tc);

        ln_kernel<<<MTOK, 256, 0, stream>>>(x, n2s + l * DD, n2b + l * DD, n2);

        // h = sum_n tc_n * (n2 @ A_n)   [MTOK,R]
        dim3 gR(RR / 64, MTOK / 128);
        for (int n = 0; n < NBASIS; ++n)
            gemm_wmma_f16<0><<<gR, 256, 0, stream>>>(n2, bA + ((size_t)l * NBASIS + n) * DD * RR,
                                                     nullptr, tc + n, NBASIS, hb,
                                                     MTOK, RR, DD, n > 0);
        // hf = sum_n tc_n * (h @ B_n)   [MTOK,DF]
        dim3 gF(DFF / 64, MTOK / 128);
        for (int n = 0; n < NBASIS; ++n)
            gemm_wmma_f16<0><<<gF, 256, 0, stream>>>(hb, bBm + ((size_t)l * NBASIS + n) * RR * DFF,
                                                     nullptr, tc + n, NBASIS, hf,
                                                     MTOK, DFF, RR, n > 0);

        gelu_kernel<<<(MTOK * DFF) / 256, 256, 0, stream>>>(hf);

        // x += hf @ dww + dwb
        gemm_wmma_f16<0><<<gDD, 256, 0, stream>>>(hf, dww + (size_t)l * DFF * DD, dwb + l * DD,
                                                  nullptr, 0, x, MTOK, DD, DFF, 1);
    }

    // Final LN + tied head: out = LN(x) @ token_emb^T   [MTOK, V]
    ln_kernel<<<MTOK, 256, 0, stream>>>(x, nfs, nfb, n1);
    gemm_wmma_f16<1><<<dim3(VV / 64, MTOK / 128), 256, 0, stream>>>(n1, te, nullptr, nullptr, 0,
                                                                    out, MTOK, VV, DD, 0);
}
